// TransformerV1_65472481460312
// MI455X (gfx1250) — matrix operand order repaired and checked by reading, not compiled, not run on hardware
//
#include <hip/hip_runtime.h>
#include <hip/hip_bf16.h>

// ---------------------------------------------------------------------------
// Transformer encoder-decoder forward for MI455X (gfx1250), wave32 + WMMA.
// GEMMs: v_wmma_f32_16x16x32_bf16, block tile 64x128, K-stage 64, wave tile
// 32x32 (8 WMMAs per stage), register-staged software pipeline so global
// loads for stage k+1 overlap the WMMAs of stage k. f32->bf16 staging uses
// v_perm_b32 packing (1 perm + 2 adds per pair).
// Problem sizes: V=16384 E=768 R=4 L=5 B=8 S=1024.
// ---------------------------------------------------------------------------

#define TV 16384
#define TE 768
#define TR 4
#define TL 5
#define TB 8
#define TS 1024
#define TM (TB * TS)          // 8192 token rows
#define THID (TR * TE)        // 3072

#define BLK_M 64
#define BLK_N 128
#define BLK_K 64
#define KPAD 72               // LDS row stride (elems): 16B-aligned frags, few bank conflicts

typedef __attribute__((ext_vector_type(16))) __bf16 v16bf;
typedef __attribute__((ext_vector_type(8)))  float  v8f;

__device__ __forceinline__ unsigned short f2bf_bits(float f) {
  union { float f; unsigned u; } a; a.f = f;
  return (unsigned short)((a.u + 0x8000u) >> 16);    // round-half-up
}
// pack bf16(lo), bf16(hi) into one dword: 2 adds + 1 v_perm_b32
__device__ __forceinline__ unsigned f2bf_pk(float lo, float hi) {
  union { float f; unsigned u; } a, b; a.f = lo; b.f = hi;
  return __builtin_amdgcn_perm(b.u + 0x8000u, a.u + 0x8000u, 0x07060302u);
}

// ---------------------------------------------------------------------------
// WMMA GEMM: C[b] = act(A[b] (MxK) * W[b] (KxN) + bias)
// 8 waves as 2(M) x 4(N); wave tile 32x32 in four v8f accumulators.
// TRANS_B: W stored [N,K] row-major (used for Q @ K^T).
// Pipeline: [stage s to LDS] barrier [issue loads s+1 | 8 WMMAs on s] barrier
// ---------------------------------------------------------------------------
template <bool TRANS_B, bool RELU>
__global__ __launch_bounds__(256)
void gemm_wmma_bf16(const float* __restrict__ A, const float* __restrict__ W,
                    const float* __restrict__ bias, float* __restrict__ C,
                    int K, int lda, int ldb, int ldc,
                    long strideA, long strideB, long strideC) {
  const int bz = blockIdx.z;
  A += (long)bz * strideA;
  W += (long)bz * strideB;
  C += (long)bz * strideC;

  const int m0 = blockIdx.y * BLK_M;
  const int n0 = blockIdx.x * BLK_N;
  const int tid  = threadIdx.x;
  const int wave = tid >> 5;
  const int lane = tid & 31;
  const int wm = wave >> 2;       // 0..1  -> m offset wm*32
  const int wn = wave & 3;        // 0..3  -> n offset wn*32
  const int half = lane >> 4;     // 0..1
  const int l16  = lane & 15;

  __shared__ unsigned short sA[BLK_M * KPAD];   // [m][k] k-contiguous
  __shared__ unsigned short sB[BLK_N * KPAD];   // [n][k] k-contiguous

  // staging assignments: fully contiguous 128B/64B per-thread global reads
  const int arow = tid >> 2;               // A row 0..63
  const int akq  = (tid & 3) * 16;         // A k offset 0/16/32/48
  const int brow = TRANS_B ? (tid >> 1) : (tid >> 2);   // n (0..127) | k (0..63)
  const int bq   = TRANS_B ? ((tid & 1) * 32)           // k offset
                           : ((tid & 3) * 32);          // n offset

  union { float4 v[4]; float f[16]; } ra;   // prefetched A slab
  union { float4 v[8]; float f[32]; } rb;   // prefetched W slab

  auto g_loadA = [&](int k0) {
    const float* s = A + (long)(m0 + arow) * lda + (k0 + akq);
#pragma unroll
    for (int j = 0; j < 4; ++j) ra.v[j] = *(const float4*)(s + 4 * j);
  };
  auto g_loadB = [&](int k0) {
    const float* s = TRANS_B ? (W + (long)(n0 + brow) * ldb + (k0 + bq))
                             : (W + (long)(k0 + brow) * ldb + (n0 + bq));
#pragma unroll
    for (int j = 0; j < 8; ++j) rb.v[j] = *(const float4*)(s + 4 * j);
  };
  auto stage = [&]() {
#pragma unroll
    for (int j = 0; j < 8; ++j)   // A: 16 vals -> 8 packed dwords (b128-able)
      *(unsigned*)&sA[arow * KPAD + akq + 2 * j] = f2bf_pk(ra.f[2*j], ra.f[2*j+1]);
    if (TRANS_B) {
#pragma unroll
      for (int j = 0; j < 16; ++j)
        *(unsigned*)&sB[brow * KPAD + bq + 2 * j] = f2bf_pk(rb.f[2*j], rb.f[2*j+1]);
    } else {
#pragma unroll
      for (int j = 0; j < 32; ++j)  // transpose into n-major LDS
        sB[(bq + j) * KPAD + brow] = f2bf_bits(rb.f[j]);
    }
  };

  v8f acc00 = {0.f,0.f,0.f,0.f,0.f,0.f,0.f,0.f};
  v8f acc01 = acc00, acc10 = acc00, acc11 = acc00;

  const int ml0 = wm * 32 + l16, ml1 = ml0 + 16;
  const int nl0 = wn * 32 + l16, nl1 = nl0 + 16;

  g_loadA(0); g_loadB(0);
  for (int k0 = 0; k0 < K; k0 += BLK_K) {
    stage();                       // waits on in-flight loads, converts, LDS store
    __syncthreads();
    if (k0 + BLK_K < K) {          // issue next-stage loads; latency hidden by WMMAs
      g_loadA(k0 + BLK_K);
      g_loadB(k0 + BLK_K);
      if (k0 + 2 * BLK_K < K)      // L2 hint two stages ahead (global_prefetch_b8)
        __builtin_prefetch(A + (long)(m0 + arow) * lda + (k0 + 2 * BLK_K + akq), 0, 0);
    }
    // ---- 2 sub-K steps x 4 WMMAs, fragments in contiguous k order (same for A and B) ---------
#pragma unroll
    for (int kk = 0; kk < BLK_K; kk += 32) {
      union { v16bf v; unsigned u[8]; } fa0, fa1, fb0, fb1;
#pragma unroll
      for (int r = 0; r < 8; ++r) {
        // A 16x32: VGPR r holds K = half*16 + 2r, +1
        const int ka = kk + half * 16 + 2 * r;
        fa0.u[r] = *(const unsigned*)&sA[ml0 * KPAD + ka];
        fa1.u[r] = *(const unsigned*)&sA[ml1 * KPAD + ka];
        // B 32x16: lane = column n, VGPR r holds K = half*16 + 2r, +1
        const int kb = kk + half * 16 + 2 * r;
        fb0.u[r] = *(const unsigned*)&sB[nl0 * KPAD + kb];
        fb1.u[r] = *(const unsigned*)&sB[nl1 * KPAD + kb];
      }
      acc00 = __builtin_amdgcn_wmma_f32_16x16x32_bf16(false, fa0.v, false, fb0.v, (short)0, acc00, false, false);
      acc01 = __builtin_amdgcn_wmma_f32_16x16x32_bf16(false, fa0.v, false, fb1.v, (short)0, acc01, false, false);
      acc10 = __builtin_amdgcn_wmma_f32_16x16x32_bf16(false, fa1.v, false, fb0.v, (short)0, acc10, false, false);
      acc11 = __builtin_amdgcn_wmma_f32_16x16x32_bf16(false, fa1.v, false, fb1.v, (short)0, acc11, false, false);
    }
    __syncthreads();               // LDS reads done before next stage() rewrites
  }

  // ---- epilogue: C/D layout M = r + 8*half, N = lane&15 --------------------
  union { v8f v; float f[8]; } u00, u01, u10, u11;
  u00.v = acc00; u01.v = acc01; u10.v = acc10; u11.v = acc11;
  const int rb2 = m0 + wm * 32 + half * 8;    // + mi*16 + r
  const int cb  = n0 + wn * 32 + l16;         // + ni*16
  const float bv0 = bias ? bias[cb]      : 0.f;
  const float bv1 = bias ? bias[cb + 16] : 0.f;
#pragma unroll
  for (int r = 0; r < 8; ++r) {
    float v00 = u00.f[r] + bv0, v01 = u01.f[r] + bv1;
    float v10 = u10.f[r] + bv0, v11 = u11.f[r] + bv1;
    if (RELU) {
      v00 = fmaxf(v00, 0.f); v01 = fmaxf(v01, 0.f);
      v10 = fmaxf(v10, 0.f); v11 = fmaxf(v11, 0.f);
    }
    C[(long)(rb2 + r)      * ldc + cb]      = v00;
    C[(long)(rb2 + r)      * ldc + cb + 16] = v01;
    C[(long)(rb2 + r + 16) * ldc + cb]      = v10;
    C[(long)(rb2 + r + 16) * ldc + cb + 16] = v11;
  }
}

// ---------------------------------------------------------------------------
// Embedding gather + sinusoidal positional encoding
// ---------------------------------------------------------------------------
__global__ __launch_bounds__(256)
void embed_posenc_kernel(const int* __restrict__ seq,
                         const float* __restrict__ embed,
                         float* __restrict__ x) {
  const int row = blockIdx.x;            // 0..B*S-1
  const int s   = row % TS;
  const int tok = seq[row];
  const float neg_log_base = -9.210340371976184f / (float)TE;  // -ln(10000)/E
  for (int e = threadIdx.x; e < TE; e += 256) {
    const int i = e >> 1;
    const float div = __expf((float)(2 * i) * neg_log_base);
    const float ang = (float)s * div;
    const float pe  = (e & 1) ? __cosf(ang) : __sinf(ang);
    x[(long)row * TE + e] = embed[(long)tok * TE + e] + pe;
  }
}

// ---------------------------------------------------------------------------
// Row softmax over scores [B*S rows, S cols], in place, scale + optional
// causal mask (col <= row-within-seq valid).
// ---------------------------------------------------------------------------
__global__ __launch_bounds__(256)
void softmax_kernel(float* __restrict__ sc, float scale, int causal) {
  const int row = blockIdx.x;
  const int s   = row % TS;
  float* p = sc + (long)row * TS;
  __shared__ float red[256];
  const int tid = threadIdx.x;

  float mx = -3.4e38f;
  for (int c = tid; c < TS; c += 256) {
    const bool ok = (!causal) || (c <= s);
    const float v = ok ? p[c] * scale : -3.4e38f;
    mx = fmaxf(mx, v);
  }
  red[tid] = mx; __syncthreads();
  for (int o = 128; o > 0; o >>= 1) {
    if (tid < o) red[tid] = fmaxf(red[tid], red[tid + o]);
    __syncthreads();
  }
  mx = red[0]; __syncthreads();

  float sum = 0.f;
  for (int c = tid; c < TS; c += 256) {
    const bool ok = (!causal) || (c <= s);
    const float e = ok ? __expf(p[c] * scale - mx) : 0.f;
    p[c] = e; sum += e;
  }
  red[tid] = sum; __syncthreads();
  for (int o = 128; o > 0; o >>= 1) {
    if (tid < o) red[tid] += red[tid + o];
    __syncthreads();
  }
  const float inv = 1.f / red[0];
  for (int c = tid; c < TS; c += 256) p[c] *= inv;
}

// ---------------------------------------------------------------------------
// out[row] = LayerNorm(in1[row] (+ in2[row])) * g + b   (population variance)
// out may alias in1/in2 (row buffered in LDS).
// ---------------------------------------------------------------------------
__global__ __launch_bounds__(256)
void layernorm_kernel(const float* __restrict__ in1,
                      const float* __restrict__ in2,
                      const float* __restrict__ g,
                      const float* __restrict__ b,
                      float* __restrict__ out) {
  const int row = blockIdx.x;
  const float* x1 = in1 + (long)row * TE;
  const float* x2 = in2 ? in2 + (long)row * TE : nullptr;
  float* o = out + (long)row * TE;
  __shared__ float xs[TE];
  __shared__ float red[256];
  const int tid = threadIdx.x;

  float s = 0.f;
  for (int e = tid; e < TE; e += 256) {
    float v = x1[e] + (x2 ? x2[e] : 0.f);
    xs[e] = v; s += v;
  }
  red[tid] = s; __syncthreads();
  for (int o2 = 128; o2 > 0; o2 >>= 1) {
    if (tid < o2) red[tid] += red[tid + o2];
    __syncthreads();
  }
  const float mean = red[0] / (float)TE; __syncthreads();

  float vs = 0.f;
  for (int e = tid; e < TE; e += 256) {
    const float d = xs[e] - mean; vs += d * d;
  }
  red[tid] = vs; __syncthreads();
  for (int o2 = 128; o2 > 0; o2 >>= 1) {
    if (tid < o2) red[tid] += red[tid + o2];
    __syncthreads();
  }
  const float inv = rsqrtf(red[0] / (float)TE + 1e-5f);
  for (int e = tid; e < TE; e += 256)
    o[e] = (xs[e] - mean) * inv * g[e] + b[e];
}

// ---------------------------------------------------------------------------
// Final: out[b, v] = y[b, S-1, :] @ unembed[:, v]   (tiny: 8 x 16384 x 768)
// ---------------------------------------------------------------------------
__global__ __launch_bounds__(256)
void unembed_kernel(const float* __restrict__ y,
                    const float* __restrict__ Wu,
                    float* __restrict__ out) {
  const int v = blockIdx.x * 256 + threadIdx.x;
  const int b = blockIdx.y;
  const float* yr = y + ((long)b * TS + (TS - 1)) * TE;
  float acc = 0.f;
  for (int e = 0; e < TE; ++e)
    acc = fmaf(yr[e], Wu[(long)e * TV + v], acc);
  out[(long)b * TV + v] = acc;
}

// ---------------------------------------------------------------------------
// Host orchestration
// ---------------------------------------------------------------------------
extern "C" void kernel_launch(void* const* d_in, const int* in_sizes, int n_in,
                              void* d_out, int out_size, void* d_ws, size_t ws_size,
                              hipStream_t stream) {
  if (n_in < 39) return;

  const int*   seq     = (const int*)d_in[0];
  const float* embed   = (const float*)d_in[1];
  const float* unembed = (const float*)d_in[2];
  // encoder params (each leading dim L)
  const float *e_wq = (const float*)d_in[3],  *e_bq = (const float*)d_in[4];
  const float *e_wk = (const float*)d_in[5],  *e_bk = (const float*)d_in[6];
  const float *e_wv = (const float*)d_in[7],  *e_bv = (const float*)d_in[8];
  const float *e_g1 = (const float*)d_in[9],  *e_b1n = (const float*)d_in[10];
  const float *e_g2 = (const float*)d_in[11], *e_b2n = (const float*)d_in[12];
  const float *e_w1 = (const float*)d_in[13], *e_bb1 = (const float*)d_in[14];
  const float *e_w2 = (const float*)d_in[15], *e_bb2 = (const float*)d_in[16];
  // decoder params
  const float *d_wq  = (const float*)d_in[17], *d_bq  = (const float*)d_in[18];
  const float *d_wk  = (const float*)d_in[19], *d_bk  = (const float*)d_in[20];
  const float *d_wv  = (const float*)d_in[21], *d_bv  = (const float*)d_in[22];
  const float *d_wq2 = (const float*)d_in[23], *d_bq2 = (const float*)d_in[24];
  const float *d_wk2 = (const float*)d_in[25], *d_bk2 = (const float*)d_in[26];
  const float *d_wv2 = (const float*)d_in[27], *d_bv2 = (const float*)d_in[28];
  const float *d_g1  = (const float*)d_in[29], *d_b1n = (const float*)d_in[30];
  const float *d_g2  = (const float*)d_in[31], *d_b2n = (const float*)d_in[32];
  const float *d_g3  = (const float*)d_in[33], *d_b3n = (const float*)d_in[34];
  const float *d_w1  = (const float*)d_in[35], *d_bb1 = (const float*)d_in[36];
  const float *d_w2  = (const float*)d_in[37], *d_bb2 = (const float*)d_in[38];

  // ---- workspace carve (all f32) ------------------------------------------
  float* ws = (float*)d_ws;
  size_t off = 0;
  auto carve = [&](size_t n) { float* p = ws + off; off += (n + 63) & ~(size_t)63; return p; };
  float* X   = carve((size_t)TM * TE);         // posenc input / decoder state
  float* H   = carve((size_t)TM * TE);         // encoder state
  float* Q   = carve((size_t)TM * TE);
  float* Kb  = carve((size_t)TM * TE);
  float* Vb  = carve((size_t)TM * TE);
  float* ATT = carve((size_t)TM * TE);
  float* TMP = carve((size_t)TM * TE);
  float* SC  = carve((size_t)TB * TS * TS);    // attention scores / probs
  float* HID = carve((size_t)TM * THID);       // MLP hidden
  float* KVS = carve((size_t)TL * TM * TE);    // encoder per-layer outputs
  if (off * sizeof(float) > ws_size) return;   // defensive: scratch too small

  const float SCALE = 0.03608439182435161f;    // 1/sqrt(768)

  auto gemm = [&](const float* A, const float* W, const float* bias, float* C,
                  int m, int n, int k, int lda, int ldb, int ldc,
                  long sa, long sb, long sc, int batches,
                  bool transB, bool relu) {
    dim3 grid(n / BLK_N, m / BLK_M, batches), block(256);
    if (transB)
      gemm_wmma_bf16<true,  false><<<grid, block, 0, stream>>>(A, W, bias, C, k, lda, ldb, ldc, sa, sb, sc);
    else if (relu)
      gemm_wmma_bf16<false, true ><<<grid, block, 0, stream>>>(A, W, bias, C, k, lda, ldb, ldc, sa, sb, sc);
    else
      gemm_wmma_bf16<false, false><<<grid, block, 0, stream>>>(A, W, bias, C, k, lda, ldb, ldc, sa, sb, sc);
  };

  // ---- embedding + positional encoding ------------------------------------
  embed_posenc_kernel<<<TM, 256, 0, stream>>>(seq, embed, X);
  hipMemcpyAsync(H, X, (size_t)TM * TE * sizeof(float),
                 hipMemcpyDeviceToDevice, stream);

  const long WEE = (long)TE * TE, WEH = (long)TE * THID;

  // ---- encoder -------------------------------------------------------------
  for (int i = 0; i < TL; ++i) {
    gemm(H, e_wq + i * WEE, e_bq + i * TE, Q,  TM, TE, TE, TE, TE, TE, 0, 0, 0, 1, false, false);
    gemm(H, e_wk + i * WEE, e_bk + i * TE, Kb, TM, TE, TE, TE, TE, TE, 0, 0, 0, 1, false, false);
    gemm(H, e_wv + i * WEE, e_bv + i * TE, Vb, TM, TE, TE, TE, TE, TE, 0, 0, 0, 1, false, false);
    // scores = Q @ K^T  (batched over B)
    gemm(Q, Kb, nullptr, SC, TS, TS, TE, TE, TE, TS,
         (long)TS * TE, (long)TS * TE, (long)TS * TS, TB, true, false);
    softmax_kernel<<<TM, 256, 0, stream>>>(SC, SCALE, 0);
    // attn = P @ V
    gemm(SC, Vb, nullptr, ATT, TS, TE, TS, TS, TE, TE,
         (long)TS * TS, (long)TS * TE, (long)TS * TE, TB, false, false);
    layernorm_kernel<<<TM, 256, 0, stream>>>(H, ATT, e_g1 + i * TE, e_b1n + i * TE, H);
    // MLP: relu(relu(H w1 + b1) w2 + b2)
    gemm(H,   e_w1 + i * WEH, e_bb1 + i * THID, HID, TM, THID, TE, TE, THID, THID, 0, 0, 0, 1, false, true);
    gemm(HID, e_w2 + i * WEH, e_bb2 + i * TE,   TMP, TM, TE, THID, THID, TE, TE, 0, 0, 0, 1, false, true);
    layernorm_kernel<<<TM, 256, 0, stream>>>(H, TMP, e_g2 + i * TE, e_b2n + i * TE, H);
    hipMemcpyAsync(KVS + (size_t)i * TM * TE, H, (size_t)TM * TE * sizeof(float),
                   hipMemcpyDeviceToDevice, stream);
  }

  // ---- decoder (state lives in X) -----------------------------------------
  float* Y = X;
  for (int i = 0; i < TL; ++i) {
    // masked self-attention
    gemm(Y, d_wq + i * WEE, d_bq + i * TE, Q,  TM, TE, TE, TE, TE, TE, 0, 0, 0, 1, false, false);
    gemm(Y, d_wk + i * WEE, d_bk + i * TE, Kb, TM, TE, TE, TE, TE, TE, 0, 0, 0, 1, false, false);
    gemm(Y, d_wv + i * WEE, d_bv + i * TE, Vb, TM, TE, TE, TE, TE, TE, 0, 0, 0, 1, false, false);
    gemm(Q, Kb, nullptr, SC, TS, TS, TE, TE, TE, TS,
         (long)TS * TE, (long)TS * TE, (long)TS * TS, TB, true, false);
    softmax_kernel<<<TM, 256, 0, stream>>>(SC, SCALE, 1);
    gemm(SC, Vb, nullptr, ATT, TS, TE, TS, TS, TE, TE,
         (long)TS * TS, (long)TS * TE, (long)TS * TE, TB, false, false);
    layernorm_kernel<<<TM, 256, 0, stream>>>(Y, ATT, d_g1 + i * TE, d_b1n + i * TE, Y);
    // cross-attention against encoder layer-i output
    const float* KV = KVS + (size_t)i * TM * TE;
    gemm(Y,  d_wq2 + i * WEE, d_bq2 + i * TE, Q,  TM, TE, TE, TE, TE, TE, 0, 0, 0, 1, false, false);
    gemm(KV, d_wk2 + i * WEE, d_bk2 + i * TE, Kb, TM, TE, TE, TE, TE, TE, 0, 0, 0, 1, false, false);
    gemm(KV, d_wv2 + i * WEE, d_bv2 + i * TE, Vb, TM, TE, TE, TE, TE, TE, 0, 0, 0, 1, false, false);
    gemm(Q, Kb, nullptr, SC, TS, TS, TE, TE, TE, TS,
         (long)TS * TE, (long)TS * TE, (long)TS * TS, TB, true, false);
    softmax_kernel<<<TM, 256, 0, stream>>>(SC, SCALE, 0);
    gemm(SC, Vb, nullptr, ATT, TS, TE, TS, TS, TE, TE,
         (long)TS * TS, (long)TS * TE, (long)TS * TE, TB, false, false);
    layernorm_kernel<<<TM, 256, 0, stream>>>(Y, ATT, d_g2 + i * TE, d_b2n + i * TE, Y);
    // MLP + LN
    gemm(Y,   d_w1 + i * WEH, d_bb1 + i * THID, HID, TM, THID, TE, TE, THID, THID, 0, 0, 0, 1, false, true);
    gemm(HID, d_w2 + i * WEH, d_bb2 + i * TE,   TMP, TM, TE, THID, THID, TE, TE, 0, 0, 0, 1, false, true);
    layernorm_kernel<<<TM, 256, 0, stream>>>(Y, TMP, d_g3 + i * TE, d_b3n + i * TE, Y);
  }

  // ---- final projection of last position ----------------------------------
  unembed_kernel<<<dim3(TV / 256, TB), 256, 0, stream>>>(Y, unembed, (float*)d_out);
  (void)in_sizes; (void)out_size;
}
